// NestedMoEModel_22625887715743
// MI455X (gfx1250) — matrix operand order repaired and checked by reading, not compiled, not run on hardware
//
#include <hip/hip_runtime.h>

// Problem constants (reference: B,D,H,G,E = 32768,256,256,8,8)
#define BB 32768
#define DD 256
#define HH 256
#define GG 8
#define EE 8
#define GH 2048   // G*H
#define GE 64     // G*E

typedef float  v2f  __attribute__((ext_vector_type(2)));
typedef float  v4f  __attribute__((ext_vector_type(4)));
typedef float  v8f  __attribute__((ext_vector_type(8)));
typedef __bf16 v4bf __attribute__((ext_vector_type(4)));
typedef __bf16 v8bf __attribute__((ext_vector_type(8)));
typedef __bf16 v16bf __attribute__((ext_vector_type(16)));

// ---------------------------------------------------------------------------
// Kernel 1: Wsum[g*H+h][d] = sum_e W_exp[g][e][h][d];  bsum[gh] = sum_e b_exp
// ---------------------------------------------------------------------------
__global__ void moe_wsum_kernel(const float* __restrict__ W_exp,
                                const float* __restrict__ b_exp,
                                float* __restrict__ Wsum,
                                float* __restrict__ bsum) {
    int idx = blockIdx.x * 256 + threadIdx.x;   // 2048*256 threads
    int gh = idx >> 8;          // 0..2047
    int d  = idx & 255;
    int g  = gh >> 8;           // H == 256
    int h  = gh & 255;
    const float* p = W_exp + ((size_t)(g * EE) * HH + h) * DD + d;
    float acc = 0.f;
#pragma unroll
    for (int e = 0; e < EE; ++e) acc += p[(size_t)e * HH * DD];
    Wsum[(size_t)gh * DD + d] = acc;
    if (d == 0) {
        float bs = 0.f;
        const float* bp = b_exp + (size_t)(g * EE) * HH + h;
#pragma unroll
        for (int e = 0; e < EE; ++e) bs += bp[(size_t)e * HH];
        bsum[gh] = bs;
    }
}

// ---------------------------------------------------------------------------
// Kernel 2: s[b] = sum(softmax(x[b] @ W_gate^T + b_gate)) computed faithfully.
// One wave32 per row; wave reductions via shfl_xor.
// ---------------------------------------------------------------------------
__global__ void moe_gate_kernel(const float* __restrict__ x,
                                const float* __restrict__ W_gate,
                                const float* __restrict__ b_gate,
                                float* __restrict__ s) {
    const int lane = threadIdx.x & 31;
    const int wave = threadIdx.x >> 5;
    const int b = blockIdx.x * 8 + wave;        // 4096 blocks * 8 waves
    const float* xr = x + (size_t)b * DD;

    v4f xv0 = *(const v4f*)(xr + lane * 8);
    v4f xv1 = *(const v4f*)(xr + lane * 8 + 4);

    float l0 = 0.f, l1 = 0.f;   // logits for experts `lane` and `lane+32`
#pragma unroll 4
    for (int e = 0; e < GE; ++e) {
        const float* wr = W_gate + (size_t)e * DD + lane * 8;
        v4f w0 = *(const v4f*)(wr);
        v4f w1 = *(const v4f*)(wr + 4);
        float p = xv0.x * w0.x + xv0.y * w0.y + xv0.z * w0.z + xv0.w * w0.w +
                  xv1.x * w1.x + xv1.y * w1.y + xv1.z * w1.z + xv1.w * w1.w;
#pragma unroll
        for (int off = 16; off > 0; off >>= 1) p += __shfl_xor(p, off, 32);
        p += b_gate[e];
        if ((e & 31) == lane) { if (e < 32) l0 = p; else l1 = p; }
    }
    float m = fmaxf(l0, l1);
#pragma unroll
    for (int off = 16; off > 0; off >>= 1) m = fmaxf(m, __shfl_xor(m, off, 32));
    float e0 = __expf(l0 - m);
    float e1 = __expf(l1 - m);
    float den = e0 + e1;
#pragma unroll
    for (int off = 16; off > 0; off >>= 1) den += __shfl_xor(den, off, 32);
    float sm = e0 / den + e1 / den;
#pragma unroll
    for (int off = 16; off > 0; off >>= 1) sm += __shfl_xor(sm, off, 32);
    if (lane == 0) s[b] = sm;
}

// ---------------------------------------------------------------------------
// Kernel 3: out[b][n] = s[b] * ( x[b,:] . Wsum[n,:] + bsum[n] )
// Split-precision bf16x3 WMMA (x = xh+xl, W = Wh+Wl; acc += xh*Wh + xh*Wl
// + xl*Wh), f32 accumulate: ~f32 accuracy at bf16 matrix rate.
// Block: 256 threads = 8 waves, tile M=128 x N=64, K in two 128-phases.
// ---------------------------------------------------------------------------
#define LDS_STRIDE 136   // bf16 elems; 272 B = 68 dwords == 4 mod 64 -> 2-way perfect

__global__ void __launch_bounds__(256)
moe_gemm_kernel(const float* __restrict__ x,
                const float* __restrict__ Wsum,
                const float* __restrict__ bsum,
                const float* __restrict__ s,
                float* __restrict__ out) {
    __shared__ __bf16 ldsH[64 * LDS_STRIDE];   // 17,408 B
    __shared__ __bf16 ldsL[64 * LDS_STRIDE];   // 17,408 B

    const int tid  = threadIdx.x;
    const int lane = tid & 31;
    const int wave = tid >> 5;
    const int hi   = lane >> 4;       // 0: lanes 0-15, 1: lanes 16-31
    const int nl   = lane & 15;

    const int row0  = blockIdx.x * 128 + wave * 16;   // this wave's 16 rows
    const int nbase = blockIdx.y * 64;                // block's 64 columns

    // 16-bit A layout: lanes 0-15 (M=0..15) hold K {k..k+7, k+16..k+23};
    // lanes 16-31 (M=0..15) hold K {k+8..k+15, k+24..k+31}.
    const float* xrow = x + (size_t)(row0 + nl) * DD;
    const int hi8 = hi * 8;

    v8f acc[4] = {};   // 4 tiles of 16x16 -> wave covers 16x64

#pragma unroll
    for (int p = 0; p < 2; ++p) {
        // ---- stage Wsum[nbase..+63][p*128..+127] into LDS as bf16 hi/lo ----
#pragma unroll
        for (int pass = 0; pass < 8; ++pass) {
            int r  = pass * 8 + (tid >> 5);
            int c4 = (tid & 31) * 4;
            v4f w = *(const v4f*)(Wsum + (size_t)(nbase + r) * DD + p * 128 + c4);
            v4bf hv, lv;
#pragma unroll
            for (int j = 0; j < 4; ++j) {
                float f = w[j];
                __bf16 h = (__bf16)f;
                hv[j] = h;
                lv[j] = (__bf16)(f - (float)h);
            }
            *(v4bf*)&ldsH[r * LDS_STRIDE + c4] = hv;
            *(v4bf*)&ldsL[r * LDS_STRIDE + c4] = lv;
        }
        __syncthreads();

        if (p == 0) __builtin_prefetch(xrow + 128, 0, 0);  // next k-phase of A

        // ---- K loop: 4 steps of K=32 ----
#pragma unroll
        for (int kk = 0; kk < 128; kk += 32) {
            const int kg = p * 128 + kk;
            // load 16 f32 of A for this lane, split into bf16 hi/lo fragments
            v4f c0a = *(const v4f*)(xrow + kg + hi8);
            v4f c0b = *(const v4f*)(xrow + kg + hi8 + 4);
            v4f c1a = *(const v4f*)(xrow + kg + 16 + hi8);
            v4f c1b = *(const v4f*)(xrow + kg + 16 + hi8 + 4);
            v16bf ah, al;
#pragma unroll
            for (int j = 0; j < 4; ++j) {
                float f0 = c0a[j], f1 = c0b[j], f2 = c1a[j], f3 = c1b[j];
                __bf16 h0 = (__bf16)f0, h1 = (__bf16)f1, h2 = (__bf16)f2, h3 = (__bf16)f3;
                ah[j]      = h0;  al[j]      = (__bf16)(f0 - (float)h0);
                ah[4 + j]  = h1;  al[4 + j]  = (__bf16)(f1 - (float)h1);
                ah[8 + j]  = h2;  al[8 + j]  = (__bf16)(f2 - (float)h2);
                ah[12 + j] = h3;  al[12 + j] = (__bf16)(f3 - (float)h3);
            }
#pragma unroll
            for (int t = 0; t < 4; ++t) {
                // 16-bit B layout: N = lane&15; lanes 0-15 hold K {kk..kk+7, kk+16..kk+23},
                // lanes 16-31 hold K {kk+8..kk+15, kk+24..kk+31} (matches A).
                const int base = (t * 16 + nl) * LDS_STRIDE + kk + hi * 8;
                v8bf bh0 = *(const v8bf*)&ldsH[base];
                v8bf bh1 = *(const v8bf*)&ldsH[base + 16];
                v8bf bl0 = *(const v8bf*)&ldsL[base];
                v8bf bl1 = *(const v8bf*)&ldsL[base + 16];
                v16bf bh = __builtin_shufflevector(bh0, bh1, 0,1,2,3,4,5,6,7,
                                                   8,9,10,11,12,13,14,15);
                v16bf bl = __builtin_shufflevector(bl0, bl1, 0,1,2,3,4,5,6,7,
                                                   8,9,10,11,12,13,14,15);
                acc[t] = __builtin_amdgcn_wmma_f32_16x16x32_bf16(
                    false, ah, false, bh, (short)0, acc[t], false, false);
                acc[t] = __builtin_amdgcn_wmma_f32_16x16x32_bf16(
                    false, ah, false, bl, (short)0, acc[t], false, false);
                acc[t] = __builtin_amdgcn_wmma_f32_16x16x32_bf16(
                    false, al, false, bh, (short)0, acc[t], false, false);
            }
        }
        __syncthreads();
    }

    // ---- epilogue: out = s[b] * (acc + bsum[n]) ----
    // C/D layout: VGPR r -> (M=r, N=lane) lanes 0-15 ; (M=8+r, N=lane-16) lanes 16-31
    float sv[8];
#pragma unroll
    for (int r = 0; r < 8; ++r) sv[r] = s[row0 + r + hi * 8];

#pragma unroll
    for (int t = 0; t < 4; ++t) {
        const int col = nbase + t * 16 + nl;
        const float bv = bsum[col];
#pragma unroll
        for (int r = 0; r < 8; ++r) {
            out[(size_t)(row0 + r + hi * 8) * GH + col] = sv[r] * (acc[t][r] + bv);
        }
    }
}

// ---------------------------------------------------------------------------
// Launch
// ---------------------------------------------------------------------------
extern "C" void kernel_launch(void* const* d_in, const int* in_sizes, int n_in,
                              void* d_out, int out_size, void* d_ws, size_t ws_size,
                              hipStream_t stream) {
    const float* x      = (const float*)d_in[0];   // [B, D]
    const float* W_gate = (const float*)d_in[1];   // [G*E, D]
    const float* b_gate = (const float*)d_in[2];   // [G*E]
    const float* W_exp  = (const float*)d_in[3];   // [G, E, H, D]
    const float* b_exp  = (const float*)d_in[4];   // [G, E, H]
    float* out = (float*)d_out;                    // [B, G*H]

    // workspace layout
    float* Wsum = (float*)d_ws;                    // [2048, 256]
    float* bsum = Wsum + (size_t)GH * DD;          // [2048]
    float* s    = bsum + GH;                       // [B]

    moe_wsum_kernel<<<(GH * DD) / 256, 256, 0, stream>>>(W_exp, b_exp, Wsum, bsum);
    moe_gate_kernel<<<BB / 8, 256, 0, stream>>>(x, W_gate, b_gate, s);

    dim3 grid(BB / 128, GH / 64);
    moe_gemm_kernel<<<grid, 256, 0, stream>>>(x, Wsum, bsum, s, out);
}
